// CGDN_14439680049290
// MI455X (gfx1250) — matrix operand order repaired and checked by reading, not compiled, not run on hardware
//
#include <hip/hip_runtime.h>

// ---------------- types for WMMA ----------------
typedef __attribute__((ext_vector_type(16))) _Float16 v16h;
typedef __attribute__((ext_vector_type(8)))  _Float16 v8h;
typedef __attribute__((ext_vector_type(8)))  float    v8f;

#define NN   50000
#define EE   640000
#define PP   1024
#define LL   4
#define HIDD 128
#define MITER 8          // M-tiles per wave in the GEMM (B frags reused 8x)

__device__ __forceinline__ float gelu_f(float x) {
    return 0.5f * x * (1.0f + erff(x * 0.70710678118654752f));
}

__device__ __forceinline__ float wave_sum(float v) {
    #pragma unroll
    for (int off = 16; off; off >>= 1) v += __shfl_xor(v, off, 32);
    return v;
}

__device__ __forceinline__ void atomicMaxFloat(float* addr, float val) {
    // monotonic bit trick: valid when destination is initialized to -inf
    if (val >= 0.0f) atomicMax((int*)addr, __float_as_int(val));
    else             atomicMin((unsigned int*)addr, (unsigned int)__float_as_int(val));
}

// ---------------- generic fill / f32->f16 convert ----------------
__global__ void fill_kernel(float* __restrict__ p, float v, int n) {
    int i = blockIdx.x * blockDim.x + threadIdx.x;
    if (i < n) p[i] = v;
}

__global__ void cvt_f16_kernel(const float* __restrict__ in, _Float16* __restrict__ o, int n) {
    int i = blockIdx.x * blockDim.x + threadIdx.x;
    if (i < n) o[i] = (_Float16)in[i];
}

// ---------------- pack weights into WMMA B-fragment order ----------------
// Bp[((nt*4+kt)*32+lane)*16 + i] = f16( B[(kt*32 + (lane>>4)*16 + i)*ncols + nt*16 + (lane&15)] )
// (B 32x16 fragment, contiguous k order chosen here: lanes0-15 hold K=kb..kb+15, lanes16-31 K=kb+16..kb+31)
__global__ void pack_b_kernel(const float* __restrict__ B, _Float16* __restrict__ Bp, int ncols) {
    int idx = blockIdx.x * blockDim.x + threadIdx.x;
    int total = (ncols >> 4) * 4 * 32;
    if (idx >= total) return;
    int l  = idx & 31;
    int kt = (idx >> 5) & 3;
    int nt = idx >> 7;
    int n  = nt * 16 + (l & 15);
    int kb = kt * 32 + ((l >> 4) << 4);
    _Float16* o = Bp + (size_t)idx * 16;
    #pragma unroll
    for (int i = 0; i < 16; ++i) o[i] = (_Float16)B[(size_t)(kb + i) * ncols + n];
}

// ---------------- WMMA GEMM: C[M x ncols] = A16[M x 128] * B + bias ----------------
// One wave: 1 N-tile, MITER consecutive M-tiles. B fragments live in registers
// and are reused across all M-tiles. All 4 k-tiles' A fragments are issued
// before the WMMA chain so global loads pipeline instead of serializing.
__global__ void gemm128_wmma(const _Float16* __restrict__ A16,
                             const _Float16* __restrict__ Bp,
                             const float* __restrict__ bias,
                             float* __restrict__ C,
                             int ncols, int mtiles, int act) {
    const int lane  = threadIdx.x & 31;
    const int wid   = threadIdx.x >> 5;
    const int ntile = blockIdx.y * (blockDim.x >> 5) + wid;
    const int mbase = blockIdx.x * MITER;
    const int lr    = lane & 15;
    const int hi    = lane >> 4;

    // B fragments (packed, 32B-aligned contiguous per lane)
    v16h b[4];
    #pragma unroll
    for (int kt = 0; kt < 4; ++kt)
        b[kt] = *(const v16h*)(Bp + (((size_t)ntile * 4 + kt) * 32 + lane) * 16);

    const int cn = ntile * 16 + lr;
    const float bv = bias ? bias[cn] : 0.0f;

    for (int mi = 0; mi < MITER; ++mi) {
        const int mt = mbase + mi;
        if (mt >= mtiles) return;          // wave-uniform: EXEC stays all-1s
        const int row0 = mt * 16;
        const _Float16* arow = A16 + (size_t)(row0 + lr) * 128;
        if (mt + 1 < mtiles)               // warm L2/WGP$ for next M-tile's A rows
            __builtin_prefetch(A16 + (size_t)(row0 + 16 + lr) * 128, 0, 0);

        // issue all A-fragment loads for this M-tile up front
        v16h a[4];
        #pragma unroll
        for (int kt = 0; kt < 4; ++kt) {
            // A k order used here (contiguous): lanes0-15 K=0..15 ; lanes16-31 K=16..31 (matches packed B)
            v8h lo = *(const v8h*)(arow + kt * 32 + (hi << 4));
            v8h hh = *(const v8h*)(arow + kt * 32 + (hi << 4) + 8);
            a[kt] = __builtin_shufflevector(lo, hh, 0, 1, 2, 3, 4, 5, 6, 7,
                                            8, 9, 10, 11, 12, 13, 14, 15);
        }
        v8f acc = {};
        #pragma unroll
        for (int kt = 0; kt < 4; ++kt)
            acc = __builtin_amdgcn_wmma_f32_16x16x32_f16(
                false, a[kt], false, b[kt], (short)0, acc, false, false);

        #pragma unroll
        for (int v = 0; v < 8; ++v) {
            // C/D layout: VGPR v -> M = v (lanes0-15) / v+8 (lanes16-31), N = lane&15
            int m = row0 + v + (hi << 3);
            float r = acc[v] + bv;
            if (act) r = gelu_f(r);
            C[(size_t)m * ncols + cn] = r;
        }
    }
}

// ---------------- preprocessing: degree + loop attr ----------------
__global__ void deg_kernel(const int* __restrict__ ei, const float* __restrict__ eattr,
                           float* __restrict__ deg, float* __restrict__ lsum) {
    int e = blockIdx.x * blockDim.x + threadIdx.x;
    if (e >= EE) return;
    int t = ei[EE + e];
    atomicAdd(&deg[t], 1.0f);
    #pragma unroll
    for (int k = 0; k < 4; ++k) atomicAdd(&lsum[(size_t)t * 4 + k], eattr[(size_t)e * 4 + k]);
}

__global__ void loop_div_kernel(const float* __restrict__ lsum, const float* __restrict__ deg,
                                float* __restrict__ loopat) {
    int i = blockIdx.x * blockDim.x + threadIdx.x;
    if (i >= NN * 4) return;
    int node = i >> 2;
    loopat[i] = lsum[i] / fmaxf(deg[node], 1.0f);
}

// ---------------- encoder: h = gelu(ln(x@W+b)) ; wave per node ----------------
__global__ void encoder_kernel(const float* __restrict__ x, const float* __restrict__ W,
                               const float* __restrict__ b, const float* __restrict__ g,
                               const float* __restrict__ be, float* __restrict__ h) {
    int wid = threadIdx.x >> 5, lane = threadIdx.x & 31;
    int node = blockIdx.x * (blockDim.x >> 5) + wid;
    if (node >= NN) return;
    float xv[8];
    #pragma unroll
    for (int k = 0; k < 8; ++k) xv[k] = x[(size_t)node * 8 + k];
    float v[4]; float s = 0.0f;
    #pragma unroll
    for (int j = 0; j < 4; ++j) {
        int ch = j * 32 + lane;
        float acc = b[ch];
        #pragma unroll
        for (int k = 0; k < 8; ++k) acc += xv[k] * W[k * 128 + ch];
        v[j] = acc; s += acc;
    }
    float mean = wave_sum(s) * (1.0f / 128.0f);
    float q = 0.0f;
    #pragma unroll
    for (int j = 0; j < 4; ++j) { float d = v[j] - mean; q += d * d; }
    float rstd = rsqrtf(wave_sum(q) * (1.0f / 128.0f) + 1e-5f);
    #pragma unroll
    for (int j = 0; j < 4; ++j) {
        int ch = j * 32 + lane;
        float y = (v[j] - mean) * rstd * g[ch] + be[ch];
        h[(size_t)node * 128 + ch] = gelu_f(y);
    }
}

// ---------------- FiLM: gamma/beta from target_mp ----------------
__global__ void film_kernel(const float* __restrict__ tmp, const float* __restrict__ W1,
                            const float* __restrict__ b1, const float* __restrict__ W2,
                            const float* __restrict__ b2, float* __restrict__ gb) {
    __shared__ float g1[64];
    int t = threadIdx.x;
    float tv = tmp[0] * 1e-6f;
    if (t < 64) g1[t] = gelu_f(tv * W1[t] + b1[t]);
    __syncthreads();
    float acc = b2[t];
    #pragma unroll 4
    for (int j = 0; j < 64; ++j) acc += g1[j] * W2[j * 256 + t];
    gb[t] = (t < 128) ? (1.0f + acc) : acc;   // gamma | beta
}

// ---------------- GAT pass A: per-(edge,head) wave -> score + atomic max ----------------
__global__ void edge_score(const int* __restrict__ ei, const float* __restrict__ eattr,
                           const float* __restrict__ loopat, const float* __restrict__ xl,
                           const float* __restrict__ xr, const float* __restrict__ We,
                           const float* __restrict__ att, float* __restrict__ sc,
                           float* __restrict__ smax) {
    int wid = threadIdx.x >> 5, lane = threadIdx.x & 31;
    long gid = (long)blockIdx.x * (blockDim.x >> 5) + wid;
    const long E2 = (long)EE + NN;
    if (gid >= E2 * 4) return;
    int e  = (int)(gid >> 2);
    int hh = (int)(gid & 3);
    int s, t; const float* ea;
    if (e < EE) { s = ei[e]; t = ei[EE + e]; ea = eattr + (size_t)e * 4; }
    else        { s = t = e - EE;            ea = loopat + (size_t)s * 4; }
    int ch = hh * 32 + lane;
    float m = xl[(size_t)s * 128 + ch] + xr[(size_t)t * 128 + ch]
            + ea[0] * We[ch] + ea[1] * We[128 + ch]
            + ea[2] * We[256 + ch] + ea[3] * We[384 + ch];
    m = (m > 0.0f) ? m : 0.2f * m;           // leaky_relu(0.2)
    float p = wave_sum(m * att[ch]);
    if (lane == 0) {
        sc[(size_t)e * 4 + hh] = p;
        atomicMaxFloat(&smax[(size_t)t * 4 + hh], p);
    }
}

// ---------------- GAT pass B: exp + atomic sum ----------------
__global__ void edge_soft(const int* __restrict__ ei, const float* __restrict__ sc,
                          const float* __restrict__ smax, float* __restrict__ ab,
                          float* __restrict__ ssum) {
    long gid = (long)blockIdx.x * blockDim.x + threadIdx.x;
    const long E2 = (long)EE + NN;
    if (gid >= E2 * 4) return;
    int e = (int)(gid >> 2), hh = (int)(gid & 3);
    int t = (e < EE) ? ei[EE + e] : (e - EE);
    float a = expf(sc[(size_t)e * 4 + hh] - smax[(size_t)t * 4 + hh]);
    ab[(size_t)e * 4 + hh] = a;
    atomicAdd(&ssum[(size_t)t * 4 + hh], a);
}

// ---------------- GAT pass C: weighted scatter-add ----------------
__global__ void edge_aggr(const int* __restrict__ ei, const float* __restrict__ ab,
                          const float* __restrict__ ssum, const float* __restrict__ xl,
                          float* __restrict__ hc) {
    int e  = blockIdx.x;
    int ch = threadIdx.x;       // 128
    int hh = ch >> 5;
    int s, t;
    if (e < EE) { s = ei[e]; t = ei[EE + e]; } else { s = t = e - EE; }
    float w = ab[(size_t)e * 4 + hh] / (ssum[(size_t)t * 4 + hh] + 1e-16f);
    atomicAdd(&hc[(size_t)t * 128 + ch], w * xl[(size_t)s * 128 + ch]);
}

// ---------------- post: h = gelu(gamma*ln(hc+conv_b)+beta) + h ; wave per node ----------------
__global__ void post_ln(const float* __restrict__ hc, const float* __restrict__ convb,
                        const float* __restrict__ lng, const float* __restrict__ lnb,
                        const float* __restrict__ gb, float* __restrict__ h) {
    int wid = threadIdx.x >> 5, lane = threadIdx.x & 31;
    int node = blockIdx.x * (blockDim.x >> 5) + wid;
    if (node >= NN) return;
    float v[4]; float s = 0.0f;
    #pragma unroll
    for (int j = 0; j < 4; ++j) {
        int ch = j * 32 + lane;
        float a = hc[(size_t)node * 128 + ch] + convb[ch];
        v[j] = a; s += a;
    }
    float mean = wave_sum(s) * (1.0f / 128.0f);
    float q = 0.0f;
    #pragma unroll
    for (int j = 0; j < 4; ++j) { float d = v[j] - mean; q += d * d; }
    float rstd = rsqrtf(wave_sum(q) * (1.0f / 128.0f) + 1e-5f);
    #pragma unroll
    for (int j = 0; j < 4; ++j) {
        int ch = j * 32 + lane;
        float y = (v[j] - mean) * rstd * lng[ch] + lnb[ch];
        float z = gelu_f(gb[ch] * y + gb[128 + ch]) + h[(size_t)node * 128 + ch];
        h[(size_t)node * 128 + ch] = z;
    }
}

// ---------------- decoder tail: dc = d1@W2+b2, clip, mask, nc ----------------
__global__ void decode_kernel(const float* __restrict__ d1, const float* __restrict__ W2,
                              const float* __restrict__ b2, const float* __restrict__ x,
                              const unsigned char* __restrict__ fx,
                              const unsigned char* __restrict__ fy,
                              float* __restrict__ out) {
    int node = blockIdx.x * blockDim.x + threadIdx.x;
    if (node >= NN) return;
    float a0 = b2[0], a1 = b2[1];
    const float* r = d1 + (size_t)node * 64;
    #pragma unroll 8
    for (int j = 0; j < 64; ++j) { float v = r[j]; a0 += v * W2[j * 2]; a1 += v * W2[j * 2 + 1]; }
    a0 = fminf(fmaxf(a0, -50.0f), 50.0f);
    a1 = fminf(fmaxf(a1, -50.0f), 50.0f);
    float dx = fx[node] ? 0.0f : a0;
    float dy = fy[node] ? 0.0f : a1;
    out[(size_t)2 * NN + node * 2 + 0] = dx;             // delta
    out[(size_t)2 * NN + node * 2 + 1] = dy;
    out[(size_t)node * 2 + 0] = x[(size_t)node * 8 + 0] + dx;   // nc
    out[(size_t)node * 2 + 1] = x[(size_t)node * 8 + 1] + dy;
}

__global__ void join_mid(const int* __restrict__ jp, const float* __restrict__ nc,
                         float* __restrict__ mid) {
    int p = blockIdx.x * blockDim.x + threadIdx.x;
    if (p >= PP) return;
    int u = jp[p * 2], v = jp[p * 2 + 1];
    mid[p * 2 + 0] = 0.5f * (nc[(size_t)u * 2 + 0] + nc[(size_t)v * 2 + 0]);
    mid[p * 2 + 1] = 0.5f * (nc[(size_t)u * 2 + 1] + nc[(size_t)v * 2 + 1]);
}

__global__ void join_set(const int* __restrict__ jp, const float* __restrict__ mid,
                         float* __restrict__ nc) {
    int p = blockIdx.x * blockDim.x + threadIdx.x;
    if (p >= PP) return;
    int u = jp[p * 2], v = jp[p * 2 + 1];
    nc[(size_t)u * 2 + 0] = mid[p * 2 + 0];
    nc[(size_t)u * 2 + 1] = mid[p * 2 + 1];
    nc[(size_t)v * 2 + 0] = mid[p * 2 + 0];
    nc[(size_t)v * 2 + 1] = mid[p * 2 + 1];
}

// ---------------- host launch ----------------
extern "C" void kernel_launch(void* const* d_in, const int* in_sizes, int n_in,
                              void* d_out, int out_size, void* d_ws, size_t ws_size,
                              hipStream_t stream) {
    (void)in_sizes; (void)n_in; (void)out_size; (void)ws_size;
    const float* x        = (const float*)d_in[0];
    const int*   ei       = (const int*)d_in[1];
    const float* eattr    = (const float*)d_in[2];
    const float* tmp      = (const float*)d_in[3];
    const unsigned char* fx = (const unsigned char*)d_in[4];
    const unsigned char* fy = (const unsigned char*)d_in[5];
    const int*   jp       = (const int*)d_in[6];
    const float* enc_W    = (const float*)d_in[7];
    const float* enc_b    = (const float*)d_in[8];
    const float* enc_g    = (const float*)d_in[9];
    const float* enc_lb   = (const float*)d_in[10];
    const float* film_W1  = (const float*)d_in[11];
    const float* film_b1  = (const float*)d_in[12];
    const float* film_W2  = (const float*)d_in[13];
    const float* film_b2  = (const float*)d_in[14];
    const float* Wl       = (const float*)d_in[15];
    const float* bl       = (const float*)d_in[16];
    const float* Wr       = (const float*)d_in[17];
    const float* br       = (const float*)d_in[18];
    const float* We       = (const float*)d_in[19];
    const float* att      = (const float*)d_in[20];
    const float* conv_b   = (const float*)d_in[21];
    const float* ln_g     = (const float*)d_in[22];
    const float* ln_b     = (const float*)d_in[23];
    const float* dec_W1   = (const float*)d_in[24];
    const float* dec_b1   = (const float*)d_in[25];
    const float* dec_W2   = (const float*)d_in[26];
    const float* dec_b2   = (const float*)d_in[27];
    float* out = (float*)d_out;

    const long E2 = (long)EE + NN;
    const int  MT = NN / 16;                 // 3125 M-tiles
    const int  GX = (MT + MITER - 1) / MITER;

    // workspace carve-out (256B aligned)
    char* base = (char*)d_ws;
    auto alloc = [&](size_t bytes) -> char* {
        char* p = base;
        base += (bytes + 255) & ~(size_t)255;
        return p;
    };
    float*    h      = (float*)alloc((size_t)NN * 128 * 4);
    float*    hc     = (float*)alloc((size_t)NN * 128 * 4);
    float*    xl     = (float*)alloc((size_t)NN * 128 * 4);
    float*    xr     = (float*)alloc((size_t)NN * 128 * 4);
    float*    sc     = (float*)alloc((size_t)E2 * 4 * 4);
    float*    ab     = (float*)alloc((size_t)E2 * 4 * 4);
    float*    smax   = (float*)alloc((size_t)NN * 4 * 4);
    float*    ssum   = (float*)alloc((size_t)NN * 4 * 4);
    float*    deg    = (float*)alloc((size_t)NN * 4);
    float*    lsum   = (float*)alloc((size_t)NN * 4 * 4);
    float*    loopat = (float*)alloc((size_t)NN * 4 * 4);
    float*    gb     = (float*)alloc(256 * 4);
    float*    d1     = (float*)alloc((size_t)NN * 64 * 4);
    float*    mid    = (float*)alloc((size_t)PP * 2 * 4);
    _Float16* h16    = (_Float16*)alloc((size_t)NN * 128 * 2);
    _Float16* Wlp    = (_Float16*)alloc((size_t)LL * 128 * 128 * 2);
    _Float16* Wrp    = (_Float16*)alloc((size_t)LL * 128 * 128 * 2);
    _Float16* W1p    = (_Float16*)alloc((size_t)128 * 64 * 2);

    // pack all weight matrices into WMMA fragment order (f16), once per call
    for (int i = 0; i < LL; ++i) {
        pack_b_kernel<<<(1024 + 255) / 256, 256, 0, stream>>>(
            Wl + (size_t)i * 128 * 128, Wlp + (size_t)i * 128 * 128, 128);
        pack_b_kernel<<<(1024 + 255) / 256, 256, 0, stream>>>(
            Wr + (size_t)i * 128 * 128, Wrp + (size_t)i * 128 * 128, 128);
    }
    pack_b_kernel<<<(512 + 255) / 256, 256, 0, stream>>>(dec_W1, W1p, 64);

    // preprocessing
    fill_kernel<<<(NN + 255) / 256, 256, 0, stream>>>(deg, 0.0f, NN);
    fill_kernel<<<(NN * 4 + 255) / 256, 256, 0, stream>>>(lsum, 0.0f, NN * 4);
    deg_kernel<<<(EE + 255) / 256, 256, 0, stream>>>(ei, eattr, deg, lsum);
    loop_div_kernel<<<(NN * 4 + 255) / 256, 256, 0, stream>>>(lsum, deg, loopat);

    // encoder (wave per node, 8 waves/block)
    encoder_kernel<<<(NN + 7) / 8, 256, 0, stream>>>(x, enc_W, enc_b, enc_g, enc_lb, h);

    const float NEG_INF = -__builtin_huge_valf();
    for (int i = 0; i < LL; ++i) {
        film_kernel<<<1, 256, 0, stream>>>(tmp, film_W1 + i * 64, film_b1 + i * 64,
                                           film_W2 + (size_t)i * 64 * 256, film_b2 + i * 256, gb);
        // h -> f16 once; feeds both GEMMs
        cvt_f16_kernel<<<(NN * 128 + 255) / 256, 256, 0, stream>>>(h, h16, NN * 128);
        gemm128_wmma<<<dim3(GX, 2), 128, 0, stream>>>(
            h16, Wlp + (size_t)i * 128 * 128, bl + i * 128, xl, 128, MT, 0);
        gemm128_wmma<<<dim3(GX, 2), 128, 0, stream>>>(
            h16, Wrp + (size_t)i * 128 * 128, br + i * 128, xr, 128, MT, 0);

        fill_kernel<<<(NN * 4 + 255) / 256, 256, 0, stream>>>(smax, NEG_INF, NN * 4);
        fill_kernel<<<(NN * 4 + 255) / 256, 256, 0, stream>>>(ssum, 0.0f, NN * 4);
        fill_kernel<<<(NN * 128 + 255) / 256, 256, 0, stream>>>(hc, 0.0f, NN * 128);

        int nwaves = (int)((E2 * 4 + 7) / 8);
        edge_score<<<nwaves, 256, 0, stream>>>(ei, eattr, loopat, xl, xr,
                                               We + (size_t)i * 512, att + i * 128, sc, smax);
        edge_soft<<<(int)((E2 * 4 + 255) / 256), 256, 0, stream>>>(ei, sc, smax, ab, ssum);
        edge_aggr<<<(int)E2, 128, 0, stream>>>(ei, ab, ssum, xl, hc);

        post_ln<<<(NN + 7) / 8, 256, 0, stream>>>(hc, conv_b + i * 128, ln_g + i * 128,
                                                  ln_b + i * 128, gb, h);
    }

    // decoder: d1 = gelu(h@dec_W1+b1) via WMMA, then scalar tail + join pairs
    cvt_f16_kernel<<<(NN * 128 + 255) / 256, 256, 0, stream>>>(h, h16, NN * 128);
    gemm128_wmma<<<dim3(GX, 1), 128, 0, stream>>>(h16, W1p, dec_b1, d1, 64, MT, 1);
    decode_kernel<<<(NN + 255) / 256, 256, 0, stream>>>(d1, dec_W2, dec_b2, x, fx, fy, out);
    join_mid<<<(PP + 255) / 256, 256, 0, stream>>>(jp, out, mid);
    join_set<<<(PP + 255) / 256, 256, 0, stream>>>(jp, mid, out);
}
